// SupervisedContrastiveLoss_71889162600778
// MI455X (gfx1250) — compile-verified, matrix operand order checked by reading, not run on hardware
//
#include <hip/hip_runtime.h>
#include <math.h>

#define NROWS 8192
#define DIM   256
#define KCHUNK 32
#define NCHUNKS (DIM / KCHUNK)   // 8

typedef __attribute__((ext_vector_type(4)))  float  v4f;
typedef __attribute__((ext_vector_type(8)))  float  v8f;
typedef __attribute__((ext_vector_type(8)))  __bf16 v8bf;
typedef __attribute__((ext_vector_type(16))) __bf16 v16bf;

static __device__ __forceinline__ v16bf cat16(v8bf a, v8bf b) {
    return __builtin_shufflevector(a, b, 0, 1, 2, 3, 4, 5, 6, 7,
                                         8, 9, 10, 11, 12, 13, 14, 15);
}

// ---------------------------------------------------------------------------
// ws layout (needs ~8.1 MB):
//   [0]        double sums[2]   (pos_sum, neg_sum)
//   [16]       uint  cnts[2]    (pos_cnt, neg_cnt)
//   [256]      float norms[8192]                 (32 KB)
//   [65536]    __bf16 Ehi[8192*256]              (4 MB)
//   [4259840]  __bf16 Elo[8192*256]              (4 MB)
// ---------------------------------------------------------------------------
#define WS_NORMS 256
#define WS_EHI   65536
#define WS_ELO   4259840

__global__ void scl_init_kernel(double* sums, unsigned int* cnts) {
    if (threadIdx.x == 0) {
        sums[0] = 0.0;
        sums[1] = 0.0;
        cnts[0] = 0u;
        cnts[1] = 0u;
    }
}

// ---------------------------------------------------------------------------
// Prep: row norms + bf16 hi/lo split. One wave32 per row; each lane owns
// 8 consecutive floats -> one v8bf hi store + one v8bf lo store (b128 each).
// ---------------------------------------------------------------------------
__global__ void scl_prep_kernel(const float* __restrict__ E,
                                float* __restrict__ norms,
                                __bf16* __restrict__ Ehi,
                                __bf16* __restrict__ Elo) {
    const int wave = threadIdx.x >> 5;
    const int lane = threadIdx.x & 31;
    const int row  = blockIdx.x * 8 + wave;
    const size_t base = (size_t)row * DIM + lane * 8;

    v4f x0 = *(const v4f*)(E + base);
    v4f x1 = *(const v4f*)(E + base + 4);

    float xs[8] = {x0.x, x0.y, x0.z, x0.w, x1.x, x1.y, x1.z, x1.w};
    v8bf hv, lv;
    float s = 0.0f;
    #pragma unroll
    for (int i = 0; i < 8; ++i) {
        const float x = xs[i];
        const __bf16 h = (__bf16)x;
        hv[i] = h;
        lv[i] = (__bf16)(x - (float)h);
        s += x * x;
    }
    *(v8bf*)(Ehi + base) = hv;
    *(v8bf*)(Elo + base) = lv;

    #pragma unroll
    for (int off = 16; off > 0; off >>= 1)
        s += __shfl_down(s, off, 32);
    if (lane == 0) norms[row] = sqrtf(s);
}

// ---------------------------------------------------------------------------
// Main: 128x64 sim tile per block (8 waves, each a 16x64 strip, 4 accums).
// Gram via bf16x3: hi*hi + hi*lo + lo*hi on V_WMMA_F32_16X16X32_BF16.
// Strict lower triangle only (gi > gj); means are unchanged by symmetry.
//
// B (column-block) fragments are staged once into LDS in fragment order:
//   slot(hl, c, t, p) holds 32 lanes x 16B, where p selects VGPR0-3 / 4-7
//   halves of the 16-bit fragment (K +0..7 / +16..23 for lane<16, +8 for
//   lane>=16), c = K-chunk (32 wide), t = 16-col sub-tile.
// ---------------------------------------------------------------------------
#define SLOT(hl, c, t, p) ((((((hl) * NCHUNKS + (c)) * 4 + (t)) * 2 + (p)) * 32))

__launch_bounds__(256)
__global__ void scl_main_kernel(const __bf16* __restrict__ Ehi,
                                const __bf16* __restrict__ Elo,
                                const int* __restrict__ labels,
                                const float* __restrict__ norms,
                                double* __restrict__ sums,
                                unsigned int* __restrict__ cnts) {
    const int i0blk = blockIdx.y * 128;
    const int j0    = blockIdx.x * 64;
    // Strict lower triangle: block has work iff max(gi)=i0blk+127 > min(gj)=j0.
    if (i0blk + 127 <= j0) return;

    __shared__ __align__(16) __bf16 smem[2 * NCHUNKS * 4 * 2 * 32 * 8]; // 64 KB

    const int tid  = threadIdx.x;
    const int lane = tid & 31;
    const int wave = tid >> 5;
    const int lr   = lane & 15;
    const int khalf = (lane & 16) ? 8 : 0;

    // ------- stage column-block fragments (hi+lo) into LDS, once ----------
    for (int s = tid; s < 2 * NCHUNKS * 4 * 2 * 32; s += 256) {
        const int sl = s & 31;
        int rest = s >> 5;
        const int p  = rest & 1;  rest >>= 1;
        const int t  = rest & 3;  rest >>= 2;
        const int c  = rest & 7;  rest >>= 3;
        const int hl = rest;      // 0 = hi, 1 = lo
        const int row    = j0 + 16 * t + (sl & 15);
        const int kstart = c * KCHUNK + p * 16 + ((sl & 16) ? 8 : 0);
        const __bf16* src = (hl ? Elo : Ehi) + (size_t)row * DIM + kstart;
        *(v8bf*)(smem + (size_t)s * 8) = *(const v8bf*)src;
    }
    __syncthreads();

    // ------------------------- WMMA main loop -----------------------------
    const int i0 = i0blk + wave * 16;
    const size_t arow = (size_t)(i0 + lr) * DIM;

    v8f acc0 = {}, acc1 = {}, acc2 = {}, acc3 = {};

    #pragma unroll
    for (int c = 0; c < NCHUNKS; ++c) {
        const size_t ka = arow + c * KCHUNK + khalf;
        const v16bf a_hi = cat16(*(const v8bf*)(Ehi + ka),
                                 *(const v8bf*)(Ehi + ka + 16));
        const v16bf a_lo = cat16(*(const v8bf*)(Elo + ka),
                                 *(const v8bf*)(Elo + ka + 16));

        v8f* accs[4] = {&acc0, &acc1, &acc2, &acc3};
        #pragma unroll
        for (int t = 0; t < 4; ++t) {
            const v16bf b_hi =
                cat16(*(const v8bf*)(smem + (size_t)(SLOT(0, c, t, 0) + lane) * 8),
                      *(const v8bf*)(smem + (size_t)(SLOT(0, c, t, 1) + lane) * 8));
            const v16bf b_lo =
                cat16(*(const v8bf*)(smem + (size_t)(SLOT(1, c, t, 0) + lane) * 8),
                      *(const v8bf*)(smem + (size_t)(SLOT(1, c, t, 1) + lane) * 8));

            v8f acc = *accs[t];
            acc = __builtin_amdgcn_wmma_f32_16x16x32_bf16(
                false, a_hi, false, b_hi, (short)0, acc, false, false);
            acc = __builtin_amdgcn_wmma_f32_16x16x32_bf16(
                false, a_hi, false, b_lo, (short)0, acc, false, false);
            acc = __builtin_amdgcn_wmma_f32_16x16x32_bf16(
                false, a_lo, false, b_hi, (short)0, acc, false, false);
            *accs[t] = acc;
        }
    }

    // --------------------------- epilogue ----------------------------------
    float pos_sum = 0.0f, neg_sum = 0.0f;
    unsigned int pos_cnt = 0u, neg_cnt = 0u;

    const int half = lane >> 4;
    float ni[8];
    int   li[8];
    #pragma unroll
    for (int r = 0; r < 8; ++r) {
        const int gi = i0 + r + 8 * half;
        ni[r] = norms[gi];
        li[r] = labels[gi];
    }

    v8f accs[4] = {acc0, acc1, acc2, acc3};
    #pragma unroll
    for (int t = 0; t < 4; ++t) {
        const int gj = j0 + 16 * t + lr;
        const float nj = norms[gj];
        const int   lj = labels[gj];
        #pragma unroll
        for (int r = 0; r < 8; ++r) {
            const int gi = i0 + r + 8 * half;
            if (gi > gj) {                       // strict lower triangle only
                const float sim = accs[t][r] / fmaxf(ni[r] * nj, 1e-8f);
                if (li[r] == lj) {
                    pos_sum += logf(fmaxf(sim, 1e-6f));
                    pos_cnt++;
                } else {
                    neg_sum += logf(fmaxf(1.0f - sim, 1e-6f));
                    neg_cnt++;
                }
            }
        }
    }

    #pragma unroll
    for (int off = 16; off > 0; off >>= 1) {
        pos_sum += __shfl_down(pos_sum, off, 32);
        neg_sum += __shfl_down(neg_sum, off, 32);
        pos_cnt += __shfl_down(pos_cnt, off, 32);
        neg_cnt += __shfl_down(neg_cnt, off, 32);
    }

    __shared__ float        s_ps[8], s_ns[8];
    __shared__ unsigned int s_pc[8], s_nc[8];
    if (lane == 0) {
        s_ps[wave] = pos_sum;
        s_ns[wave] = neg_sum;
        s_pc[wave] = pos_cnt;
        s_nc[wave] = neg_cnt;
    }
    __syncthreads();

    if (tid == 0) {
        float ps = 0.0f, ns = 0.0f;
        unsigned int pc = 0u, nc = 0u;
        #pragma unroll
        for (int w = 0; w < 8; ++w) {
            ps += s_ps[w]; ns += s_ns[w]; pc += s_pc[w]; nc += s_nc[w];
        }
        atomicAdd(&sums[0], (double)ps);
        atomicAdd(&sums[1], (double)ns);
        atomicAdd(&cnts[0], pc);
        atomicAdd(&cnts[1], nc);
    }
}

// ---------------------------------------------------------------------------
// Finalize. Lower-triangle sums/counts are exactly half the full off-diagonal
// sums/counts (symmetry), so the 2x factors cancel inside each mean.
// ---------------------------------------------------------------------------
__global__ void scl_finalize_kernel(const double* __restrict__ sums,
                                    const unsigned int* __restrict__ cnts,
                                    float* __restrict__ out) {
    if (threadIdx.x == 0) {
        const double pl = -sums[0] / (double)cnts[0];
        const double nl = -sums[1] / (double)cnts[1];
        out[0] = (float)(pl + nl);
    }
}

extern "C" void kernel_launch(void* const* d_in, const int* in_sizes, int n_in,
                              void* d_out, int out_size, void* d_ws, size_t ws_size,
                              hipStream_t stream) {
    const float* E      = (const float*)d_in[0];   // [8192, 256] fp32
    const int*   labels = (const int*)d_in[1];     // [8192]

    double*       sums  = (double*)d_ws;
    unsigned int* cnts  = (unsigned int*)((char*)d_ws + 16);
    float*        norms = (float*)((char*)d_ws + WS_NORMS);
    __bf16*       Ehi   = (__bf16*)((char*)d_ws + WS_EHI);
    __bf16*       Elo   = (__bf16*)((char*)d_ws + WS_ELO);

    hipLaunchKernelGGL(scl_init_kernel, dim3(1), dim3(32), 0, stream,
                       sums, cnts);
    hipLaunchKernelGGL(scl_prep_kernel, dim3(NROWS / 8), dim3(256), 0, stream,
                       E, norms, Ehi, Elo);

    dim3 grid(NROWS / 64, NROWS / 128);   // (col tiles, row tiles)
    hipLaunchKernelGGL(scl_main_kernel, grid, dim3(256), 0, stream,
                       Ehi, Elo, labels, norms, sums, cnts);

    hipLaunchKernelGGL(scl_finalize_kernel, dim3(1), dim3(32), 0, stream,
                       sums, cnts, (float*)d_out);
}
